// MetaLayer_7103875907705
// MI455X (gfx1250) — matrix operand order repaired and checked by reading, not compiled, not run on hardware
//
#include <hip/hip_runtime.h>

#define N_NODES 20000
#define N_EDGES 640000
#define DD 64
#define HH 128

typedef __attribute__((ext_vector_type(16))) __bf16 v16bf;
typedef __attribute__((ext_vector_type(8)))  float  v8f;
typedef __attribute__((ext_vector_type(16))) unsigned short v16u;
typedef __attribute__((ext_vector_type(8)))  unsigned short v8u;

__device__ __forceinline__ unsigned short bf_bits(float f) {
    return __builtin_bit_cast(unsigned short, (__bf16)f);   // hw RNE convert
}

// A-fragment (16x32) from a bf16 row-major row. Per ISA 16-bit A layout:
// lane holds row r=lane&15; elems 0..7 = K(half*8 + 0..7), elems 8..15 = +16.
__device__ __forceinline__ v16bf load_a_bf(const unsigned short* p, int half) {
    v8u lo = *(const v8u*)(p + half * 8);
    v8u hi = *(const v8u*)(p + half * 8 + 16);
    v16u u;
#pragma unroll
    for (int i = 0; i < 8; ++i) { u[i] = lo[i]; u[8 + i] = hi[i]; }
    return __builtin_bit_cast(v16bf, u);
}

// A-fragment from an f32 row (converted on the fly with hw casts).
__device__ __forceinline__ v16bf load_a_f32(const float* rowptr, int half) {
    const float4* p0 = (const float4*)(rowptr + half * 8);
    const float4* p1 = (const float4*)(rowptr + half * 8 + 16);
    float4 f0 = p0[0], f1 = p0[1];
    float4 f2 = p1[0], f3 = p1[1];
    v16bf a;
    a[0]  = (__bf16)f0.x; a[1]  = (__bf16)f0.y; a[2]  = (__bf16)f0.z; a[3]  = (__bf16)f0.w;
    a[4]  = (__bf16)f1.x; a[5]  = (__bf16)f1.y; a[6]  = (__bf16)f1.z; a[7]  = (__bf16)f1.w;
    a[8]  = (__bf16)f2.x; a[9]  = (__bf16)f2.y; a[10] = (__bf16)f2.z; a[11] = (__bf16)f2.w;
    a[12] = (__bf16)f3.x; a[13] = (__bf16)f3.y; a[14] = (__bf16)f3.z; a[15] = (__bf16)f3.w;
    return a;
}

// 3x concat(64) -> 128 -> relu -> 64 MLP for one wave's 32-row tile
// (two 16-row sub-tiles sharing every B-fragment load -> 2x arithmetic
// intensity on the weight stream). afrag(sub, ks, half) yields the A
// fragment for sub-tile `sub`, K-step ks. hrow = wave-private LDS,
// 32 rows x 136-ushort pitch (conflict-free 16B accesses).
template <typename AFrag>
__device__ __forceinline__ void mlp_tile2(
    AFrag afrag,
    const unsigned short* __restrict__ W1p, const float* __restrict__ b1,
    const unsigned short* __restrict__ W2p, const float* __restrict__ b2,
    unsigned short* hrow, int lane, v8f out[2][4])
{
    const int r = lane & 15, half = lane >> 4;
    v8f acc[2][8];
#pragma unroll
    for (int s = 0; s < 2; ++s)
#pragma unroll
        for (int nt = 0; nt < 8; ++nt)
            acc[s][nt] = (v8f){0.f,0.f,0.f,0.f,0.f,0.f,0.f,0.f};

#pragma unroll
    for (int ks = 0; ks < 6; ++ks) {
        v16bf a0 = afrag(0, ks, half);
        v16bf a1 = afrag(1, ks, half);
#pragma unroll
        for (int nt = 0; nt < 8; ++nt) {
            v16bf b = *(const v16bf*)(W1p + ((ks * 8 + nt) * 32 + lane) * 16);
            acc[0][nt] = __builtin_amdgcn_wmma_f32_16x16x32_bf16(
                false, a0, false, b, (short)0, acc[0][nt], false, false);
            acc[1][nt] = __builtin_amdgcn_wmma_f32_16x16x32_bf16(
                false, a1, false, b, (short)0, acc[1][nt], false, false);
        }
    }

    // bias + relu, stage hidden as bf16 in LDS (C-layout -> row-major)
#pragma unroll
    for (int s = 0; s < 2; ++s)
#pragma unroll
        for (int nt = 0; nt < 8; ++nt) {
            float bias = b1[nt * 16 + r];
#pragma unroll
            for (int j = 0; j < 8; ++j) {
                float v = acc[s][nt][j] + bias;
                v = v > 0.f ? v : 0.f;
                hrow[(s * 16 + j + half * 8) * 136 + nt * 16 + r] = bf_bits(v);
            }
        }
    asm volatile("" ::: "memory");   // DS ops are wave-in-order; fence compiler

    v8f acc2[2][4];
#pragma unroll
    for (int s = 0; s < 2; ++s)
#pragma unroll
        for (int nt = 0; nt < 4; ++nt)
            acc2[s][nt] = (v8f){0.f,0.f,0.f,0.f,0.f,0.f,0.f,0.f};

#pragma unroll
    for (int ks = 0; ks < 4; ++ks) {
        v16bf a0 = load_a_bf(hrow + (r)      * 136 + ks * 32, half);
        v16bf a1 = load_a_bf(hrow + (16 + r) * 136 + ks * 32, half);
#pragma unroll
        for (int nt = 0; nt < 4; ++nt) {
            v16bf b = *(const v16bf*)(W2p + ((ks * 4 + nt) * 32 + lane) * 16);
            acc2[0][nt] = __builtin_amdgcn_wmma_f32_16x16x32_bf16(
                false, a0, false, b, (short)0, acc2[0][nt], false, false);
            acc2[1][nt] = __builtin_amdgcn_wmma_f32_16x16x32_bf16(
                false, a1, false, b, (short)0, acc2[1][nt], false, false);
        }
    }
#pragma unroll
    for (int s = 0; s < 2; ++s)
#pragma unroll
        for (int nt = 0; nt < 4; ++nt) {
            float bias = b2[nt * 16 + r];
#pragma unroll
            for (int j = 0; j < 8; ++j) out[s][nt][j] = acc2[s][nt][j] + bias;
        }
}

// Re-swizzle all four f32 weight matrices into bf16 B-fragment layout:
// frag = (ks, nt); element(lane, i) = W[(ks*32 + (i>>3)*16 + (lane>>4)*8 + (i&7))*ncols + nt*16 + (lane&15)]
// ws layout (ushort units): We1p[0..48frag) We2p[48..64) Wn1p[64..112) Wn2p[112..128), 512 ushort/frag
__global__ void prep_weights(const float* __restrict__ We1, const float* __restrict__ We2,
                             const float* __restrict__ Wn1, const float* __restrict__ Wn2,
                             unsigned short* __restrict__ out)
{
    int idx = blockIdx.x * blockDim.x + threadIdx.x;     // 0..65535
    int frag = idx >> 9;
    int lane = (idx >> 4) & 31;
    int i    = idx & 15;
    const float* W; int ncols; int f;
    if      (frag < 48)  { W = We1; ncols = 128; f = frag;       }
    else if (frag < 64)  { W = We2; ncols = 64;  f = frag - 48;  }
    else if (frag < 112) { W = Wn1; ncols = 128; f = frag - 64;  }
    else                 { W = Wn2; ncols = 64;  f = frag - 112; }
    int ntc = (ncols == 128) ? 8 : 4;
    int ks = f / ntc, nt = f % ntc;
    int k = ks * 32 + ((i >> 3) << 4) + ((lane >> 4) << 3) + (i & 7);
    int n = nt * 16 + (lane & 15);
    out[idx] = bf_bits(W[k * ncols + n]);
}

// Zero the scatter buffer and pre-convert the gathered tables x, f to bf16.
__global__ void prep_data(const float* __restrict__ x, const float* __restrict__ f,
                          unsigned short* __restrict__ xbf, unsigned short* __restrict__ fbf,
                          float* __restrict__ agg, int n)
{
    int i = blockIdx.x * blockDim.x + threadIdx.x;
    if (i < n) {
        xbf[i] = bf_bits(x[i]);
        fbf[i] = bf_bits(f[i]);
        agg[i] = 0.f;
    }
}

__global__ __launch_bounds__(256, 1) void edge_kernel(
    const unsigned short* __restrict__ xbf, const int* __restrict__ eidx,
    const float* __restrict__ eattr,
    const unsigned short* __restrict__ W1p, const float* __restrict__ b1,
    const unsigned short* __restrict__ W2p, const float* __restrict__ b2,
    float* __restrict__ edge_out, float* __restrict__ agg)
{
    __shared__ unsigned short hbuf[8 * 32 * 136];
    __shared__ int dbuf[8 * 32];
    const int lane = threadIdx.x & 31, w = threadIdx.x >> 5;
    const int r = lane & 15, half = lane >> 4;
    const int e0 = blockIdx.x * 256 + w * 32;   // E divisible by 256: no guard

    const unsigned short* xs[2];
    const unsigned short* xd[2];
    const float*          ea[2];
#pragma unroll
    for (int s = 0; s < 2; ++s) {
        int e = e0 + s * 16 + r;
        int srci = eidx[e];
        int dsti = eidx[N_EDGES + e];
        if (lane < 16) dbuf[w * 32 + s * 16 + r] = dsti;
        xs[s] = xbf + (size_t)srci * DD;
        xd[s] = xbf + (size_t)dsti * DD;
        ea[s] = eattr + (size_t)e * DD;
    }
    asm volatile("" ::: "memory");

    auto afrag = [&](int s, int ks, int hf) -> v16bf {
        if (ks < 2) return load_a_bf(xs[s] + ks * 32, hf);
        if (ks < 4) return load_a_bf(xd[s] + (ks - 2) * 32, hf);
        return load_a_f32(ea[s] + (ks - 4) * 32, hf);
    };

    v8f out[2][4];
    mlp_tile2(afrag, W1p, b1, W2p, b2, hbuf + w * 32 * 136, lane, out);

#pragma unroll
    for (int s = 0; s < 2; ++s)
#pragma unroll
        for (int nt = 0; nt < 4; ++nt) {
            int n = nt * 16 + r;
#pragma unroll
            for (int j = 0; j < 8; ++j) {
                int m = s * 16 + j + half * 8;
                float v = out[s][nt][j];
                edge_out[(size_t)(e0 + m) * DD + n] = v;
                atomicAdd(&agg[(size_t)dbuf[w * 32 + m] * DD + n], v);
            }
        }
}

__global__ __launch_bounds__(256, 1) void node_kernel(
    const unsigned short* __restrict__ xbf, const float* __restrict__ agg,
    const unsigned short* __restrict__ fbf,
    const unsigned short* __restrict__ W1p, const float* __restrict__ b1,
    const unsigned short* __restrict__ W2p, const float* __restrict__ b2,
    float* __restrict__ x_out)
{
    __shared__ unsigned short hbuf[8 * 32 * 136];
    const int lane = threadIdx.x & 31, w = threadIdx.x >> 5;
    const int r = lane & 15, half = lane >> 4;
    const int n0 = blockIdx.x * 256 + w * 32;

    const unsigned short* xr[2];
    const float*          ar[2];
    const unsigned short* fr[2];
#pragma unroll
    for (int s = 0; s < 2; ++s) {
        int row = n0 + s * 16 + r;
        int rowc = row < N_NODES ? row : N_NODES - 1;   // clamp loads at tail
        xr[s] = xbf + (size_t)rowc * DD;
        ar[s] = agg + (size_t)rowc * DD;
        fr[s] = fbf + (size_t)rowc * DD;
    }

    auto afrag = [&](int s, int ks, int hf) -> v16bf {
        if (ks < 2) return load_a_bf(xr[s] + ks * 32, hf);
        if (ks < 4) return load_a_f32(ar[s] + (ks - 2) * 32, hf);
        return load_a_bf(fr[s] + (ks - 4) * 32, hf);
    };

    v8f out[2][4];
    mlp_tile2(afrag, W1p, b1, W2p, b2, hbuf + w * 32 * 136, lane, out);

#pragma unroll
    for (int s = 0; s < 2; ++s)
#pragma unroll
        for (int nt = 0; nt < 4; ++nt) {
            int n = nt * 16 + r;
#pragma unroll
            for (int j = 0; j < 8; ++j) {
                int m = s * 16 + j + half * 8;
                if (n0 + m < N_NODES)
                    x_out[(size_t)(n0 + m) * DD + n] = out[s][nt][j];
            }
        }
}

extern "C" void kernel_launch(void* const* d_in, const int* in_sizes, int n_in,
                              void* d_out, int out_size, void* d_ws, size_t ws_size,
                              hipStream_t stream)
{
    const float* x     = (const float*)d_in[0];
    const int*   eidx  = (const int*)d_in[1];
    const float* eattr = (const float*)d_in[2];
    const float* f     = (const float*)d_in[3];
    const float* We1   = (const float*)d_in[4];
    const float* be1   = (const float*)d_in[5];
    const float* We2   = (const float*)d_in[6];
    const float* be2   = (const float*)d_in[7];
    const float* Wn1   = (const float*)d_in[8];
    const float* bn1   = (const float*)d_in[9];
    const float* Wn2   = (const float*)d_in[10];
    const float* bn2   = (const float*)d_in[11];

    float* x_out    = (float*)d_out;
    float* edge_out = x_out + (size_t)N_NODES * DD;

    // d_ws layout:
    //   [0, 128KB)           bf16 weight fragments
    //   [128KB, +5.12MB)     f32 agg scatter buffer
    //   then xbf (2.56MB), fbf (2.56MB)
    char* ws = (char*)d_ws;
    unsigned short* Wp  = (unsigned short*)ws;
    float*          agg = (float*)(ws + 131072);
    unsigned short* xbf = (unsigned short*)(ws + 131072 + (size_t)N_NODES * DD * 4);
    unsigned short* fbf = xbf + (size_t)N_NODES * DD;

    prep_weights<<<256, 256, 0, stream>>>(We1, We2, Wn1, Wn2, Wp);
    prep_data<<<(N_NODES * DD + 255) / 256, 256, 0, stream>>>(
        x, f, xbf, fbf, agg, N_NODES * DD);

    edge_kernel<<<N_EDGES / 256, 256, 0, stream>>>(
        xbf, eidx, eattr,
        Wp + 0,        be1,
        Wp + 48 * 512, be2,
        edge_out, agg);

    node_kernel<<<(N_NODES + 255) / 256, 256, 0, stream>>>(
        xbf, agg, fbf,
        Wp + 64 * 512,  bn1,
        Wp + 112 * 512, bn2,
        x_out);
}
